// DenoiseNet_81295140978999
// MI455X (gfx1250) — compile-verified, matrix operand order checked by reading, not run on hardware
//
#include <hip/hip_runtime.h>

typedef __attribute__((ext_vector_type(16))) __bf16        v16bf;
typedef __attribute__((ext_vector_type(8)))  float         v8f;
typedef __attribute__((ext_vector_type(4)))  unsigned int  u32x4;

#define BN_EPS 1e-5f

__device__ __forceinline__ unsigned short f2bf(float f) {
  unsigned int u = __float_as_uint(f);
  u += 0x7FFFu + ((u >> 16) & 1u);          // round-to-nearest-even
  return (unsigned short)(u >> 16);
}

// ---------------------------------------------------------------- utilities
__global__ void zero_kernel(float* __restrict__ p, int n) {
  int i = blockIdx.x * blockDim.x + threadIdx.x;
  if (i < n) p[i] = 0.f;
}

// ------------------------------------------------- prep: seeds / weights / clean
// one block per batch b; Nn == Nc == 1024
__global__ __launch_bounds__(256) void prep_kernel(
    const float* __restrict__ noisy, const float* __restrict__ cleanIn,
    const float* __restrict__ seeds,
    float* __restrict__ pin, float* __restrict__ cleanOut,
    float* __restrict__ clean_n2, float* __restrict__ sw,
    int Nn, int Nc)
{
  __shared__ float sd2[1024];
  __shared__ float red[256];
  const int b = blockIdx.x;
  const float s0 = seeds[b*3+0], s1 = seeds[b*3+1], s2 = seeds[b*3+2];

  for (int n = threadIdx.x; n < Nn; n += 256) {
    size_t i = (size_t)(b*Nn + n) * 3;
    float a0 = noisy[i+0]-s0, a1 = noisy[i+1]-s1, a2 = noisy[i+2]-s2;
    pin[i+0]=a0; pin[i+1]=a1; pin[i+2]=a2;
    sd2[n] = a0*a0 + a1*a1 + a2*a2;
  }
  for (int n = threadIdx.x; n < Nc; n += 256) {
    size_t i = (size_t)(b*Nc + n) * 3;
    float c0 = cleanIn[i+0]-s0, c1 = cleanIn[i+1]-s1, c2 = cleanIn[i+2]-s2;
    cleanOut[i+0]=c0; cleanOut[i+1]=c1; cleanOut[i+2]=c2;
    clean_n2[b*Nc+n] = c0*c0 + c1*c1 + c2*c2;
  }
  __syncthreads();
  const float inv = 9.0f / sd2[Nn-1];     // w = exp(-d2 / (d2_last/9))
  float wv[4];
  float lsum = 0.f;
  #pragma unroll
  for (int i = 0; i < 4; ++i) {
    int n = threadIdx.x + i*256;
    float w = expf(-sd2[n]*inv);
    wv[i] = w; lsum += w;
  }
  red[threadIdx.x] = lsum;
  __syncthreads();
  for (int st = 128; st > 0; st >>= 1) {
    if ((int)threadIdx.x < st) red[threadIdx.x] += red[threadIdx.x+st];
    __syncthreads();
  }
  const float invsum = 1.0f / red[0];
  #pragma unroll
  for (int i = 0; i < 4; ++i) {
    int n = threadIdx.x + i*256;
    sw[b*Nn+n] = wv[i]*invsum;
  }
}

// ------------------------------------------------- layer 0 (K=3) scalar GEMM
__global__ void gemm3_kernel(const float* __restrict__ X, const float* __restrict__ W,
                             const float* __restrict__ bias, float* __restrict__ Y,
                             int M, int N)
{
  int i = blockIdx.x*blockDim.x + threadIdx.x;
  if (i >= M*N) return;
  int m = i / N, n = i - m*N;
  const float* x = X + (size_t)m*3;
  const float* w = W + (size_t)n*3;
  Y[i] = bias[n] + x[0]*w[0] + x[1]*w[1] + x[2]*w[2];
}

// column sum / sumsq (used only for the K=3 layer)
__global__ __launch_bounds__(256) void colstats_kernel(
    const float* __restrict__ Y, float* __restrict__ ssum, float* __restrict__ ssq, int N)
{
  __shared__ float rs[256], rq[256];
  const int n   = blockIdx.x;
  const int row = blockIdx.y*256 + threadIdx.x;
  float v = Y[(size_t)row*N + n];
  rs[threadIdx.x] = v; rq[threadIdx.x] = v*v;
  __syncthreads();
  for (int st = 128; st > 0; st >>= 1) {
    if ((int)threadIdx.x < st) { rs[threadIdx.x]+=rs[threadIdx.x+st]; rq[threadIdx.x]+=rq[threadIdx.x+st]; }
    __syncthreads();
  }
  if (threadIdx.x == 0) { atomicAdd(&ssum[n], rs[0]); atomicAdd(&ssq[n], rq[0]); }
}

// ------------------------------------------------- f32 -> bf16 weight convert
__global__ void wcvt_kernel(const float* __restrict__ W, unsigned short* __restrict__ Wb, int n) {
  int i = blockIdx.x*blockDim.x + threadIdx.x;
  if (i < n) Wb[i] = f2bf(W[i]);
}

// ------------------------------------------------- bf16 WMMA GEMM + fused BN stats
// Y[M,N] = Xb[M,K] * Wb[N,K]^T + bias ; per-column sum/sumsq via atomics.
// 128 threads = 4 waves; each wave -> one 16x16 tile; grid=(N/16, M/64).
__global__ __launch_bounds__(128) void gemm_bf16_wmma_kernel(
    const unsigned short* __restrict__ Xb,
    const unsigned short* __restrict__ Wb,
    const float* __restrict__ bias,
    float* __restrict__ Y,
    float* __restrict__ ssum, float* __restrict__ ssq,
    int N, int K)
{
  const int lane  = (int)(threadIdx.x & 31u);
  const int wave  = (int)(threadIdx.x >> 5);
  const int tm    = blockIdx.y*4 + wave;     // M tile
  const int tn    = blockIdx.x;              // N tile
  const int l15   = lane & 15;
  const int khalf = (lane < 16) ? 0 : 8;     // ISA 16-bit A/B K-packing

  const unsigned short* arow = Xb + (size_t)(tm*16 + l15)*K + khalf;
  const unsigned short* brow = Wb + (size_t)(tn*16 + l15)*K + khalf;

  v8f c = {};
  for (int kk = 0; kk < K; kk += 32) {
    union { u32x4 q[2]; v16bf v; } a, bm;
    a.q[0]  = *(const u32x4*)(arow + kk);        // K = kk+khalf   .. +7
    a.q[1]  = *(const u32x4*)(arow + kk + 16);   // K = kk+16+khalf.. +7
    bm.q[0] = *(const u32x4*)(brow + kk);
    bm.q[1] = *(const u32x4*)(brow + kk + 16);
    __builtin_prefetch(arow + kk + 64, 0, 1);
    __builtin_prefetch(brow + kk + 64, 0, 1);
    c = __builtin_amdgcn_wmma_f32_16x16x32_bf16(false, a.v, false, bm.v,
                                                (short)0, c, false, false);
  }

  const int   ncol  = tn*16 + l15;
  const float bn    = bias[ncol];
  const int   rbase = tm*16 + ((lane < 16) ? 0 : 8);
  float s = 0.f, sq = 0.f;
  #pragma unroll
  for (int r = 0; r < 8; ++r) {
    float y = c[r] + bn;
    Y[(size_t)(rbase + r)*N + ncol] = y;
    s += y; sq += y*y;
  }
  atomicAdd(&ssum[ncol], s);
  atomicAdd(&ssq[ncol], sq);
}

// ------------------------------------------------- BN + ReLU, emit bf16 (+opt f32)
__global__ void bn_relu_kernel(const float* __restrict__ Y,
    const float* __restrict__ ssum, const float* __restrict__ ssq,
    const float* __restrict__ gamma, const float* __restrict__ beta,
    float* __restrict__ Xf, unsigned short* __restrict__ Xb,
    int M, int N, int writeF32)
{
  size_t i = (size_t)blockIdx.x*blockDim.x + threadIdx.x;
  if (i >= (size_t)M*N) return;
  int n = (int)(i % (size_t)N);
  float invM = 1.0f / (float)M;
  float mean = ssum[n]*invM;
  float var  = ssq[n]*invM - mean*mean;
  float v = (Y[i]-mean)*rsqrtf(var + BN_EPS)*gamma[n] + beta[n];
  v = fmaxf(v, 0.f);
  Xb[i] = f2bf(v);
  if (writeF32) Xf[i] = v;
}

// ------------------------------------------------- fc3 (256 -> 3) + tanh
__global__ void fc3_tanh_kernel(const float* __restrict__ X, const float* __restrict__ W,
                                const float* __restrict__ b, float* __restrict__ pred,
                                int M, int K)
{
  int t = blockIdx.x*blockDim.x + threadIdx.x;
  if (t >= M) return;
  const float* x = X + (size_t)t*K;
  float a0 = b[0], a1 = b[1], a2 = b[2];
  for (int k = 0; k < K; ++k) {
    float xv = x[k];
    a0 += xv*W[k]; a1 += xv*W[K+k]; a2 += xv*W[2*K+k];
  }
  pred[(size_t)t*3+0] = tanhf(a0);
  pred[(size_t)t*3+1] = tanhf(a1);
  pred[(size_t)t*3+2] = tanhf(a2);
}

// ------------------------------------------------- NN search + loss + x update
__global__ void nn_loss_kernel(float* __restrict__ pin, const float* __restrict__ clean,
    const float* __restrict__ cn2, const float* __restrict__ pred,
    const float* __restrict__ sw, float* __restrict__ loss,
    int B, int Nn, int Nc)
{
  int t = blockIdx.x*blockDim.x + threadIdx.x;
  if (t >= B*Nn) return;
  int b = t / Nn;
  float x0 = pin[(size_t)t*3+0], x1 = pin[(size_t)t*3+1], x2 = pin[(size_t)t*3+2];
  float xn2 = x0*x0 + x1*x1 + x2*x2;
  const float* cb = clean + (size_t)b*Nc*3;
  const float* c2 = cn2   + (size_t)b*Nc;
  float best = 3.4e38f; int bi = 0;
  for (int mm = 0; mm < Nc; ++mm) {
    float d = xn2 + c2[mm] - 2.f*(x0*cb[mm*3] + x1*cb[mm*3+1] + x2*cb[mm*3+2]);
    if (d < best) { best = d; bi = mm; }          // first-min, matches argmin
  }
  float n0 = cb[bi*3+0]-x0, n1 = cb[bi*3+1]-x1, n2 = cb[bi*3+2]-x2;
  float p0 = pred[(size_t)t*3+0], p1 = pred[(size_t)t*3+1], p2 = pred[(size_t)t*3+2];
  float dist = (p0-n0)*(p0-n0) + (p1-n1)*(p1-n1) + (p2-n2)*(p2-n2);
  atomicAdd(loss, sw[t]*dist*(1.0f/(float)B));
  pin[(size_t)t*3+0] = x0+p0;
  pin[(size_t)t*3+1] = x1+p1;
  pin[(size_t)t*3+2] = x2+p2;
}

// =================================================================== launcher
extern "C" void kernel_launch(void* const* d_in, const int* in_sizes, int n_in,
                              void* d_out, int out_size, void* d_ws, size_t ws_size,
                              hipStream_t stream) {
  (void)in_sizes; (void)n_in; (void)out_size; (void)ws_size;
  const int B = 32, Nn = 1024, Nc = 1024;
  const int M = B * Nn;                       // 32768 tokens

  const float* noisy   = (const float*)d_in[0];
  const float* cleanIn = (const float*)d_in[1];
  const float* seeds   = (const float*)d_in[2];
  // d_in[3] = pcl_std (unused by the reference loss)

  // ---- workspace carve-up (floats) ----
  float* ws = (float*)d_ws;
  size_t off = 0;
  float* pin      = ws + off; off += (size_t)M*3;
  float* clean    = ws + off; off += (size_t)M*3;
  float* clean_n2 = ws + off; off += (size_t)M;
  float* sw       = ws + off; off += (size_t)M;
  float* pred     = ws + off; off += (size_t)M*3;
  float* stat_sum = ws + off; off += 1024;          // stat_sum||stat_sq contiguous
  float* stat_sq  = ws + off; off += 1024;
  float* actY     = ws + off; off += (size_t)M*1024;          // GEMM output (f32)
  float* actX     = ws + off; off += (size_t)M*1024;          // f32 act (for fc3)
  unsigned short* xb = (unsigned short*)(ws + off); off += (size_t)M*1024/2; // bf16 act
  unsigned short* wb = (unsigned short*)(ws + off);           // bf16 weights (<=512K elems)

  zero_kernel<<<1, 32, 0, stream>>>((float*)d_out, 1);
  prep_kernel<<<B, 256, 0, stream>>>(noisy, cleanIn, seeds, pin, clean, clean_n2, sw, Nn, Nc);

  static const int dims[7][2] = {{3,64},{64,128},{128,256},{256,512},{512,1024},
                                 {1024,512},{512,256}};

  for (int m = 0; m < 4; ++m) {
    const int base = 4 + 30*m;                 // 5 enc * 4 + fc1(4) + fc2(4) + fc3(2)
    for (int l = 0; l < 7; ++l) {
      const int K = dims[l][0], N = dims[l][1];
      const int pi = (l < 5) ? (base + l*4) : (base + 20 + (l-5)*4);
      const float* Wp  = (const float*)d_in[pi];
      const float* bp  = (const float*)d_in[pi+1];
      const float* gp  = (const float*)d_in[pi+2];
      const float* bep = (const float*)d_in[pi+3];

      zero_kernel<<<8, 256, 0, stream>>>(stat_sum, 2048);   // zeros sum+sq

      if (l == 0) {           // K=3: scalar GEMM + separate column stats
        gemm3_kernel<<<(M*N + 255)/256, 256, 0, stream>>>(pin, Wp, bp, actY, M, N);
        colstats_kernel<<<dim3(N, M/256), 256, 0, stream>>>(actY, stat_sum, stat_sq, N);
      } else {                // bf16 WMMA path
        wcvt_kernel<<<(N*K + 255)/256, 256, 0, stream>>>(Wp, wb, N*K);
        gemm_bf16_wmma_kernel<<<dim3(N/16, M/64), 128, 0, stream>>>(
            xb, wb, bp, actY, stat_sum, stat_sq, N, K);
      }
      const int wantF32 = (l == 6) ? 1 : 0;   // only fc2 output needed in f32 (fc3 input)
      size_t tot = (size_t)M * N;
      bn_relu_kernel<<<(unsigned)((tot + 255)/256), 256, 0, stream>>>(
          actY, stat_sum, stat_sq, gp, bep, actX, xb, M, N, wantF32);
    }
    const float* W3 = (const float*)d_in[base+28];
    const float* b3 = (const float*)d_in[base+29];
    fc3_tanh_kernel<<<(M + 255)/256, 256, 0, stream>>>(actX, W3, b3, pred, M, 256);
    nn_loss_kernel<<<(M + 255)/256, 256, 0, stream>>>(
        pin, clean, clean_n2, pred, sw, (float*)d_out, B, Nn, Nc);
  }
}
